// VanillaRNN_42640435314854
// MI455X (gfx1250) — matrix operand order repaired and checked by reading, not compiled, not run on hardware
//
#include <hip/hip_runtime.h>
#include <hip/hip_bf16.h>

typedef _Float16 v16h __attribute__((ext_vector_type(16)));
typedef _Float16 v8h  __attribute__((ext_vector_type(8)));
typedef float    v8f  __attribute__((ext_vector_type(8)));

#define SEQ    512
#define HID    256
#define BAT    1024
#define NCLS   10
#define XPITCH 516   // f32 units; 516 dwords % 64 banks == 4 -> conflict-free per-column
#define HPITCH 264   // f16 units; 528 bytes = 132 dwords % 64 == 4 -> conflict-free b128

__device__ __forceinline__ float fast_tanh(float x) {
#if __has_builtin(__builtin_amdgcn_tanhf)
    return __builtin_amdgcn_tanhf(x);
#elif __has_builtin(__builtin_amdgcn_tanh_f32)
    return __builtin_amdgcn_tanh_f32(x);
#else
    float e = __expf(2.0f * x);
    return 1.0f - 2.0f / (e + 1.0f);
#endif
}

// One workgroup owns 16 batch columns for the entire 512-step recurrence.
// 8 waves x (2 M-tiles of 16 rows) cover H=256 output rows.
// Whh (f16 WMMA A-fragments) is held in VGPRs for the whole kernel.
// h ping-pongs between two LDS buffers (f16); one barrier per time step.
// B-fragment LDS loads are software-pipelined (2-deep register rotation)
// so ds latency overlaps the WMMA chain instead of serializing on dscnt 0.
__global__ __launch_bounds__(256, 1)
void VanillaRNN_42640435314854_kernel(const float* __restrict__ x,
                                      const float* __restrict__ Whx,
                                      const float* __restrict__ Whh,
                                      const float* __restrict__ Wph,
                                      const float* __restrict__ bhp,
                                      const float* __restrict__ bpp,
                                      float* __restrict__ out)
{
    __shared__ float    xl[16 * XPITCH];          // x tile: 16 cols x 512 steps
    __shared__ _Float16 hb[2][16 * HPITCH];       // double-buffered h, col-major (n, k)

    const int tid  = threadIdx.x;
    const int wave = tid >> 5;        // 0..7
    const int lane = tid & 31;
    const int n    = lane & 15;       // tile column (N index / batch col)
    const int half = lane >> 4;       // 0 or 1
    const int b0   = blockIdx.x * 16; // first batch column of this WG

    // ---- stage x tile into LDS (coalesced row loads) ----
    for (int i = tid; i < 16 * SEQ; i += 256) {
        int row = i >> 9;           // /512
        int t   = i & (SEQ - 1);
        xl[row * XPITCH + t] = x[(size_t)(b0 + row) * SEQ + t];
    }
    // ---- zero both h buffers (h0 = 0) ----
    {
        unsigned int* hz = (unsigned int*)&hb[0][0];
        for (int i = tid; i < 2 * 16 * HPITCH / 2; i += 256) hz[i] = 0u;
    }

    // ---- load Whh A-fragments into registers (f32 -> f16), plus Whx/bh rows ----
    // A fragment (16x32 f16) lane layout: m = lane%16, half = lane/16,
//   a[j]   = A[m][16*half + j],      j = 0..7
    //   a[8+j] = A[m][16*half + 8 + j]
    v16h  Afrag[2][8];
    float whxr[2][8];
    float bhr[2][8];
    const int mrow = n; // row within M tile
    #pragma unroll
    for (int i = 0; i < 2; ++i) {
        const int mt  = wave * 2 + i;          // M tile 0..15
        const int row = mt * 16 + mrow;
        const float* wr = Whh + (size_t)row * HID;
        #pragma unroll
        for (int kc = 0; kc < 8; ++kc) {
            const int kb = kc * 32 + half * 16;
            v16h a;
            #pragma unroll
            for (int j = 0; j < 8; ++j) {
                a[j]     = (_Float16)wr[kb + j];
                a[8 + j] = (_Float16)wr[kb + 8 + j];
            }
            Afrag[i][kc] = a;
        }
        const int rb = mt * 16 + half * 8;     // C-layout rows: m = 8*half + v
        #pragma unroll
        for (int v = 0; v < 8; ++v) {
            whxr[i][v] = Whx[rb + v];
            bhr[i][v]  = bhp[rb + v];
        }
    }
    __syncthreads();

    // ---- 512-step recurrence ----
    int rd = 0;
    for (int t = 0; t < SEQ; ++t) {
        const _Float16* hr = &hb[rd][0];
        _Float16*       hw = &hb[rd ^ 1][0];
        const _Float16* hbase = hr + n * HPITCH + half * 16;

        const float xv = xl[n * XPITCH + t];   // broadcast within half-wave pairs

        v8f acc[2];
        #pragma unroll
        for (int i = 0; i < 2; ++i)
            #pragma unroll
            for (int v = 0; v < 8; ++v)
                acc[i][v] = fmaf(whxr[i][v], xv, bhr[i][v]);

        // B fragment (32x16 f16) lane layout: n = lane%16,
        //   b[j] = B[16*half + j][n], j = 0..15 (contiguous K in our LDS layout)
        // 2-deep pipelined loads: issue kc+1 before consuming kc.
        v8h blo[2], bhi[2];
        blo[0] = *(const v8h*)(hbase);
        bhi[0] = *(const v8h*)(hbase + 8);
        #pragma unroll
        for (int kc = 0; kc < 8; ++kc) {
            const int cur = kc & 1;
            const int nxt = cur ^ 1;
            if (kc < 7) {
                const _Float16* bsrc = hbase + (kc + 1) * 32;
                blo[nxt] = *(const v8h*)(bsrc);
                bhi[nxt] = *(const v8h*)(bsrc + 8);
            }
            v16h bfrag;
            #pragma unroll
            for (int j = 0; j < 8; ++j) {
                bfrag[j]     = blo[cur][j];
                bfrag[8 + j] = bhi[cur][j];
            }
            acc[0] = __builtin_amdgcn_wmma_f32_16x16x32_f16(
                         false, Afrag[0][kc], false, bfrag, (short)0, acc[0], false, false);
            acc[1] = __builtin_amdgcn_wmma_f32_16x16x32_f16(
                         false, Afrag[1][kc], false, bfrag, (short)0, acc[1], false, false);
        }

        // tanh -> f16 -> store 8 contiguous rows per lane (aligned 16B, bank-clean)
        #pragma unroll
        for (int i = 0; i < 2; ++i) {
            const int mt = wave * 2 + i;
            v8h hv;
            #pragma unroll
            for (int v = 0; v < 8; ++v) hv[v] = (_Float16)fast_tanh(acc[i][v]);
            *(v8h*)(hw + n * HPITCH + mt * 16 + half * 8) = hv;
        }
        __syncthreads();
        rd ^= 1;
    }

    // ---- final projection p = Wph @ h_final + bp  (tiny: 16 cols x 10 classes) ----
    const _Float16* hf = &hb[rd][0];
    for (int i = tid; i < 16 * NCLS; i += 256) {
        const int c  = i % NCLS;
        const int nn = i / NCLS;
        float s = bpp[c];
        const float* wr = Wph + (size_t)c * HID;
        #pragma unroll 4
        for (int k = 0; k < HID; ++k)
            s = fmaf(wr[k], (float)hf[nn * HPITCH + k], s);
        out[(size_t)(b0 + nn) * NCLS + c] = s;
    }
}

extern "C" void kernel_launch(void* const* d_in, const int* in_sizes, int n_in,
                              void* d_out, int out_size, void* d_ws, size_t ws_size,
                              hipStream_t stream) {
    const float* x    = (const float*)d_in[0];
    const float* Whx  = (const float*)d_in[1];
    const float* Whh  = (const float*)d_in[2];
    const float* Wph  = (const float*)d_in[3];
    const float* bh   = (const float*)d_in[4];
    const float* bp   = (const float*)d_in[5];
    float* out = (float*)d_out;

    dim3 grid(BAT / 16);   // 64 workgroups, each owns 16 batch columns
    dim3 block(256);       // 8 wave32s
    VanillaRNN_42640435314854_kernel<<<grid, block, 0, stream>>>(
        x, Whx, Whh, Wph, bh, bp, out);
}
